// SlidingWindowAttention_65687229825741
// MI455X (gfx1250) — matrix operand order repaired and checked by reading, not compiled, not run on hardware
//
#include <hip/hip_runtime.h>
#include <hip/hip_bf16.h>

// ---------------------------------------------------------------------------
// Sliding-window attention on gfx1250 via v_wmma_f32_16x16x32_bf16,
// with async global->LDS staged, double-buffered GEMMs.
// ---------------------------------------------------------------------------

typedef __attribute__((ext_vector_type(16))) __bf16 v16bf;
typedef __attribute__((ext_vector_type(8)))  float  v8f;
typedef int v4i_vs __attribute__((vector_size(16)));   // matches builtin's V4i

#define D_MODEL 1024
#define T_SEQ   4096
#define N_HEADS 16
#define D_HEAD  64
#define WINDOW  256

#if defined(__has_builtin)
#if __has_builtin(__builtin_amdgcn_exp2f)
#define EXP2F(x) __builtin_amdgcn_exp2f(x)
#else
#define EXP2F(x) exp2f(x)
#endif
#else
#define EXP2F(x) exp2f(x)
#endif

#if defined(__has_builtin)
#if __has_builtin(__builtin_amdgcn_global_load_async_to_lds_b128)
#define HAVE_ASYNC_LDS 1
#endif
#endif
#ifndef HAVE_ASYNC_LDS
#define HAVE_ASYNC_LDS 0
#endif

#if HAVE_ASYNC_LDS
#define ASYNC_WAIT() asm volatile("s_wait_asynccnt 0x0" ::: "memory")
#else
#define ASYNC_WAIT()
#endif

// 16-byte global -> LDS copy (async DMA when available).
__device__ __forceinline__ void cp16(__bf16* ldst, const __bf16* gsrc) {
#if HAVE_ASYNC_LDS
  __builtin_amdgcn_global_load_async_to_lds_b128(
      (__attribute__((address_space(1))) v4i_vs*)gsrc,
      (__attribute__((address_space(3))) v4i_vs*)ldst,
      0, 0);
#else
  *(uint4*)ldst = *(const uint4*)gsrc;
#endif
}

__device__ __forceinline__ v8f v8f_zero() {
  v8f z = {0.f, 0.f, 0.f, 0.f, 0.f, 0.f, 0.f, 0.f};
  return z;
}

__device__ __forceinline__ v8f wmma_bf16(v16bf a, v16bf b, v8f c) {
  // (neg_a, A, neg_b, B, c_mod, C, reuse_a, reuse_b)
  return __builtin_amdgcn_wmma_f32_16x16x32_bf16(false, a, false, b,
                                                 (short)0, c, false, false);
}

// A-matrix fragment (16x32 bf16, M x K). Lane L holds row M=L%16.
// K = klo..klo+15 contiguous, klo = (L>=16)?16:0.
// `base` points at the start of this lane's 32-element K-run (its row).
__device__ __forceinline__ v16bf load_frag_a(const __bf16* base, int lane) {
  const int klo = (lane & 16) ? 16 : 0;
  union { v16bf v; uint4 u[2]; } f;
  f.u[0] = *(const uint4*)(base + klo);
  f.u[1] = *(const uint4*)(base + klo + 8);
  return f.v;
}

// B-matrix fragment (32x16 bf16, K x N). Lane L holds column N=L%16,
// K = kc..kc+15 (kc = (L>=16)?16:0). `p` points at the lane's 16 consecutive
// K elements (caller applies the n row offset and kc).
__device__ __forceinline__ v16bf load_frag_b32(const __bf16* p) {
  union { v16bf v; uint4 u[2]; } f;
  f.u[0] = *(const uint4*)(p);
  f.u[1] = *(const uint4*)(p + 8);
  return f.v;
}

// ---------------------------------------------------------------------------
// fp32 -> bf16 conversion
// ---------------------------------------------------------------------------
__global__ void swa_cvt_bf16_kernel(const float* __restrict__ src,
                                    __bf16* __restrict__ dst, int n) {
  int i = (blockIdx.x * blockDim.x + threadIdx.x) * 4;
  if (i < n) {
    float4 v = *(const float4*)(src + i);
    union { __bf16 b[4]; uint2 u; } o;
    o.b[0] = (__bf16)v.x; o.b[1] = (__bf16)v.y;
    o.b[2] = (__bf16)v.z; o.b[3] = (__bf16)v.w;
    *(uint2*)(dst + i) = o.u;
  }
}

// ---------------------------------------------------------------------------
// Generic GEMM: Y[m,n] = sum_k A[m,k] * W[n,k]   (A: MxK, W: NxK, row-major)
// Block = 128 threads (4 waves, 2x2), block tile 128(M) x 64(N).
// Per k-step of 32: A-tile (128x32, 8KB) and B-tile (64x32, 4KB) staged into
// LDS with async global->LDS copies, double buffered; waves consume via
// ds_load_b128 fragments and 8 WMMAs each.
// mode 0: Y -> bf16 row-major (ld=N)
// mode 1: Y -> bf16 transposed (Y[n*M + m])   (used for V)
// mode 2: Y -> f32 row-major                  (final output)
// ---------------------------------------------------------------------------
__device__ __forceinline__ void stage_tiles(const __bf16* __restrict__ A,
                                            const __bf16* __restrict__ W,
                                            int K, int m0b, int n0b, int kb,
                                            __bf16* lA, __bf16* lB, int tid) {
  // A tile: one 64B row per thread (4 x 16B)
  {
    const __bf16* g = A + (size_t)(m0b + tid) * K + kb;
    __bf16* l = lA + tid * 32;
#pragma unroll
    for (int c = 0; c < 4; ++c) cp16(l + c * 8, g + c * 8);
  }
  // B tile: half a 64B row per thread (2 x 16B)
  {
    const int row = tid >> 1, off = (tid & 1) * 16;
    const __bf16* g = W + (size_t)(n0b + row) * K + kb + off;
    __bf16* l = lB + row * 32 + off;
    cp16(l, g);
    cp16(l + 8, g + 8);
  }
}

__global__ __launch_bounds__(128)
void swa_gemm_bf16_kernel(const __bf16* __restrict__ A,
                          const __bf16* __restrict__ W,
                          int M, int N, int K,
                          __bf16* __restrict__ outB,
                          float* __restrict__ outF,
                          int mode) {
  __shared__ __align__(16) __bf16 lA[2][128 * 32];
  __shared__ __align__(16) __bf16 lB[2][64 * 32];

  const int tid  = threadIdx.x;
  const int lane = tid & 31;
  const int wid  = tid >> 5;
  const int wm = wid >> 1, wn = wid & 1;
  const int m0b = blockIdx.x * 128;
  const int n0b = blockIdx.y * 64;
  const int cn  = lane & 15;               // row (A) / column (B,C) index
  const int hi8 = (lane & 16) ? 8 : 0;     // C/D row offset for upper half
  const int kc  = (lane & 16) ? 16 : 0;    // B-fragment K offset

  v8f acc[4][2];
#pragma unroll
  for (int mi = 0; mi < 4; ++mi)
#pragma unroll
    for (int ni = 0; ni < 2; ++ni) acc[mi][ni] = v8f_zero();

  stage_tiles(A, W, K, m0b, n0b, 0, lA[0], lB[0], tid);

  int cur = 0;
  for (int kb = 0; kb < K; kb += 32) {
    ASYNC_WAIT();           // my async copies from the previous stage landed
    __syncthreads();        // everyone's copies landed; prior reads finished
    if (kb + 32 < K)
      stage_tiles(A, W, K, m0b, n0b, kb + 32, lA[cur ^ 1], lB[cur ^ 1], tid);

    const __bf16* la = lA[cur];
    const __bf16* lb = lB[cur];
    v16bf af[4], bfg[2];
#pragma unroll
    for (int mi = 0; mi < 4; ++mi)
      af[mi] = load_frag_a(la + (wm * 64 + mi * 16 + cn) * 32, lane);
#pragma unroll
    for (int ni = 0; ni < 2; ++ni)
      bfg[ni] = load_frag_b32(lb + (wn * 32 + ni * 16 + cn) * 32 + kc);
#pragma unroll
    for (int mi = 0; mi < 4; ++mi)
#pragma unroll
      for (int ni = 0; ni < 2; ++ni)
        acc[mi][ni] = wmma_bf16(af[mi], bfg[ni], acc[mi][ni]);
    cur ^= 1;
  }

#pragma unroll
  for (int mi = 0; mi < 4; ++mi)
#pragma unroll
    for (int ni = 0; ni < 2; ++ni)
#pragma unroll
      for (int r = 0; r < 8; ++r) {
        int m = m0b + wm * 64 + mi * 16 + r + hi8;
        int n = n0b + wn * 32 + ni * 16 + cn;
        float v = acc[mi][ni][r];
        if (mode == 0)       outB[(size_t)m * N + n] = (__bf16)v;
        else if (mode == 1)  outB[(size_t)n * M + m] = (__bf16)v;
        else                 outF[(size_t)m * N + n] = v;
      }
}

// ---------------------------------------------------------------------------
// Flash-style sliding-window attention.
// One wave per (head, 16-query tile). 4 waves / block.
// Q,K: (T, 1024) bf16 row-major.  Vt: (1024, T) bf16 (feature-major).
// Ob: (T, 1024) bf16 row-major.
// ---------------------------------------------------------------------------
__global__ __launch_bounds__(128)
void swa_attn_kernel(const __bf16* __restrict__ Qb,
                     const __bf16* __restrict__ Kb,
                     const __bf16* __restrict__ Vt,
                     __bf16* __restrict__ Ob) {
  __shared__ __align__(16) __bf16 smem[4][16 * 32];

  const int lane = threadIdx.x & 31;
  const int wid  = threadIdx.x >> 5;
  const int cn   = lane & 15;
  const int hi8  = (lane & 16) ? 8 : 0;
  const int kc   = (lane & 16) ? 16 : 0;
  const int h    = blockIdx.y;
  const int q0   = (blockIdx.x * 4 + wid) * 16;
  __bf16* P = smem[wid];

  const float scale = 0.125f;            // 1/sqrt(64)
  const float L2E   = 1.44269504f;

  // Q A-fragments (d = 0..31 and 32..63); lane's row = q0 + cn.
  const __bf16* qrow = Qb + (size_t)(q0 + cn) * D_MODEL + h * D_HEAD;
  v16bf qa0 = load_frag_a(qrow, lane);
  v16bf qa1 = load_frag_a(qrow + 32, lane);

  float m_s[8], l_s[8];
#pragma unroll
  for (int r = 0; r < 8; ++r) { m_s[r] = -1e9f; l_s[r] = 0.f; }
  v8f o0 = v8f_zero(), o1 = v8f_zero(), o2 = v8f_zero(), o3 = v8f_zero();

  int kstart = q0 - WINDOW; if (kstart < 0) kstart = 0;
  kstart &= ~31;
  const int kend = q0 + 16;

  for (int kb = kstart; kb < kend; kb += 32) {
    // ---- S = Q @ K^T for 32 keys (two 16-key N-tiles) ----
    v8f s0 = v8f_zero(), s1 = v8f_zero();
    const __bf16* kp0 = Kb + (size_t)(kb + cn) * D_MODEL + h * D_HEAD + kc;
    const __bf16* kp1 = Kb + (size_t)(kb + 16 + cn) * D_MODEL + h * D_HEAD + kc;
    s0 = wmma_bf16(qa0, load_frag_b32(kp0),      s0);
    s0 = wmma_bf16(qa1, load_frag_b32(kp0 + 32), s0);
    s1 = wmma_bf16(qa0, load_frag_b32(kp1),      s1);
    s1 = wmma_bf16(qa1, load_frag_b32(kp1 + 32), s1);

    // ---- online softmax (per-row across the 16 lanes of each half) ----
#pragma unroll
    for (int r = 0; r < 8; ++r) {
      const int i  = q0 + r + hi8;
      const int j0 = kb + cn, j1 = j0 + 16;
      float v0 = s0[r] * scale;
      float v1 = s1[r] * scale;
      if (j0 > i || j0 < i - WINDOW) v0 = -1e38f;
      if (j1 > i || j1 < i - WINDOW) v1 = -1e38f;

      float rmax = fmaxf(v0, v1);
      rmax = fmaxf(rmax, __shfl_xor(rmax, 1, 32));
      rmax = fmaxf(rmax, __shfl_xor(rmax, 2, 32));
      rmax = fmaxf(rmax, __shfl_xor(rmax, 4, 32));
      rmax = fmaxf(rmax, __shfl_xor(rmax, 8, 32));

      const float mn    = fmaxf(m_s[r], rmax);
      const float alpha = EXP2F((m_s[r] - mn) * L2E);
      const float p0    = EXP2F((v0 - mn) * L2E);
      const float p1    = EXP2F((v1 - mn) * L2E);

      float rs = p0 + p1;
      rs += __shfl_xor(rs, 1, 32);
      rs += __shfl_xor(rs, 2, 32);
      rs += __shfl_xor(rs, 4, 32);
      rs += __shfl_xor(rs, 8, 32);

      l_s[r] = l_s[r] * alpha + rs;
      m_s[r] = mn;
      o0[r] *= alpha; o1[r] *= alpha; o2[r] *= alpha; o3[r] *= alpha;

      // stage P (16x32, row-major) for the A-fragment reload
      P[(r + hi8) * 32 + cn]      = (__bf16)p0;
      P[(r + hi8) * 32 + cn + 16] = (__bf16)p1;
    }

    // intra-wave DS ops are in-order; wait for store data explicitly anyway
    asm volatile("s_wait_dscnt 0x0" ::: "memory");

    // ---- O += P @ V (K-dim = 32 keys; 4 N-tiles over d_head=64) ----
    v16bf pa = load_frag_a(P + cn * 32, lane);
    const __bf16* vb = Vt + (size_t)(h * D_HEAD + cn) * T_SEQ + kb + kc;
    o0 = wmma_bf16(pa, load_frag_b32(vb),              o0);
    o1 = wmma_bf16(pa, load_frag_b32(vb + 16 * T_SEQ), o1);
    o2 = wmma_bf16(pa, load_frag_b32(vb + 32 * T_SEQ), o2);
    o3 = wmma_bf16(pa, load_frag_b32(vb + 48 * T_SEQ), o3);
  }

  // ---- normalize and store ----
#pragma unroll
  for (int r = 0; r < 8; ++r) {
    const float inv = 1.0f / l_s[r];
    const int m = q0 + r + hi8;
    const size_t rowoff = (size_t)m * D_MODEL + h * D_HEAD;
    Ob[rowoff + cn]      = (__bf16)(o0[r] * inv);
    Ob[rowoff + 16 + cn] = (__bf16)(o1[r] * inv);
    Ob[rowoff + 32 + cn] = (__bf16)(o2[r] * inv);
    Ob[rowoff + 48 + cn] = (__bf16)(o3[r] * inv);
  }
}

// ---------------------------------------------------------------------------
extern "C" void kernel_launch(void* const* d_in, const int* in_sizes, int n_in,
                              void* d_out, int out_size, void* d_ws, size_t ws_size,
                              hipStream_t stream) {
  const float* x  = (const float*)d_in[0];
  const float* Wq = (const float*)d_in[1];
  const float* Wk = (const float*)d_in[2];
  const float* Wv = (const float*)d_in[3];
  const float* Wo = (const float*)d_in[4];
  float* out = (float*)d_out;

  char* ws = (char*)d_ws;
  __bf16* xb  = (__bf16*)(ws + ((size_t)0  << 20));  //  8 MiB: x bf16
  __bf16* wqb = (__bf16*)(ws + ((size_t)8  << 20));  //  2 MiB
  __bf16* wkb = (__bf16*)(ws + ((size_t)10 << 20));  //  2 MiB
  __bf16* wvb = (__bf16*)(ws + ((size_t)12 << 20));  //  2 MiB
  __bf16* wob = (__bf16*)(ws + ((size_t)14 << 20));  //  2 MiB
  __bf16* Qb  = (__bf16*)(ws + ((size_t)16 << 20));  //  8 MiB
  __bf16* Kb  = (__bf16*)(ws + ((size_t)24 << 20));  //  8 MiB
  __bf16* Vt  = (__bf16*)(ws + ((size_t)32 << 20));  //  8 MiB (transposed)
  __bf16* Ab  = (__bf16*)(ws + ((size_t)40 << 20));  //  8 MiB (attn out)

  const int nX = T_SEQ * D_MODEL;
  const int nW = D_MODEL * D_MODEL;

  swa_cvt_bf16_kernel<<<dim3((nX / 4 + 255) / 256), 256, 0, stream>>>(x,  xb,  nX);
  swa_cvt_bf16_kernel<<<dim3((nW / 4 + 255) / 256), 256, 0, stream>>>(Wq, wqb, nW);
  swa_cvt_bf16_kernel<<<dim3((nW / 4 + 255) / 256), 256, 0, stream>>>(Wk, wkb, nW);
  swa_cvt_bf16_kernel<<<dim3((nW / 4 + 255) / 256), 256, 0, stream>>>(Wv, wvb, nW);
  swa_cvt_bf16_kernel<<<dim3((nW / 4 + 255) / 256), 256, 0, stream>>>(Wo, wob, nW);

  dim3 gg(T_SEQ / 128, D_MODEL / 64);
  // Q = x @ Wq^T, K = x @ Wk^T (row-major bf16); V transposed to (1024, T)
  swa_gemm_bf16_kernel<<<gg, 128, 0, stream>>>(xb, wqb, T_SEQ, D_MODEL, D_MODEL, Qb, nullptr, 0);
  swa_gemm_bf16_kernel<<<gg, 128, 0, stream>>>(xb, wkb, T_SEQ, D_MODEL, D_MODEL, Kb, nullptr, 0);
  swa_gemm_bf16_kernel<<<gg, 128, 0, stream>>>(xb, wvb, T_SEQ, D_MODEL, D_MODEL, Vt, nullptr, 1);

  swa_attn_kernel<<<dim3(T_SEQ / 64, N_HEADS), 128, 0, stream>>>(Qb, Kb, Vt, Ab);

  // out = attn @ Wo^T (f32 output)
  swa_gemm_bf16_kernel<<<gg, 128, 0, stream>>>(Ab, wob, T_SEQ, D_MODEL, D_MODEL, nullptr, out, 2);
}
